// EdgePropertyPredictionModel_85366769975467
// MI455X (gfx1250) — matrix operand order repaired and checked by reading, not compiled, not run on hardware
//
#include <hip/hip_runtime.h>
#include <hip/hip_bf16.h>
#include <stdint.h>

#define NNODES 50000
#define NEDGES 800000
#define INDIM  64
#define DMODEL 128
#define NHEAD  4
#define CHEAD  32
#define FHID   512
#define NLAYER 4
#define EPSBN  1e-5f
#define SLOPE  0.2f

typedef __attribute__((ext_vector_type(16))) __bf16 v16bf;
typedef __attribute__((ext_vector_type(8)))  __bf16 v8bf;
typedef __attribute__((ext_vector_type(8)))  float  v8f;

// ---------- helpers ----------
// pack two fp32 -> two bf16 (round-to-nearest-even) in one u32
static __device__ inline unsigned pack_bf16x2(float lo, float hi) {
    unsigned ul = __float_as_uint(lo), uh = __float_as_uint(hi);
    ul = (ul + 0x7FFFu + ((ul >> 16) & 1u)) >> 16;
    uh = (uh + 0x7FFFu + ((uh >> 16) & 1u)) >> 16;
    return (ul & 0xFFFFu) | (uh << 16);
}
// order-preserving float<->uint encoding for atomicMax over signed floats
static __device__ inline unsigned enc_f(float f) {
    unsigned u = __float_as_uint(f);
    return (u & 0x80000000u) ? ~u : (u | 0x80000000u);
}
static __device__ inline float dec_f(unsigned u) {
    unsigned b = (u & 0x80000000u) ? (u & 0x7FFFFFFFu) : ~u;
    return __uint_as_float(b);
}

// ---------- WMMA GEMM: C[M,Nn] = act(A[M,K] @ B[K,Nn] + bias (+res)) ----------
// MODE 0: bias     MODE 1: bias+ReLU     MODE 2: bias+residual
// K is compile-time (64/128/512), all multiples of 32.
template<int MODE, int K>
__global__ __launch_bounds__(256)
void wmma_gemm_kernel(const float* __restrict__ A, const float* __restrict__ B,
                      const float* __restrict__ bias, const float* __restrict__ res,
                      float* __restrict__ C, int M, int Nn)
{
    // packed bf16 pairs: As[m][k/2], Bs[n][k/2]  (B transposed for contiguous lane reads)
    __shared__ unsigned As[64][16];   // 4 KB
    __shared__ unsigned Bs[32][16];   // 2 KB

    const int t    = threadIdx.x;
    const int m0   = blockIdx.x * 64;
    const int n0   = blockIdx.y * 32;
    const int wave = t >> 5, lane = t & 31;
    const int wm   = wave >> 1, wn = wave & 1;     // 4x2 waves -> 64x32 tile
    const int lm   = lane & 15, half = lane >> 4;

    v8f acc = {};

    // A staging: thread -> row r (0..63), 8 consecutive k (two float4 loads)
    const int ar   = t >> 2;
    const int ac   = (t & 3) * 8;               // k base in elements
    const int gmA  = m0 + ar;
    const bool rowok = (gmA < M);
    const float4* Arow = (const float4*)(A + (size_t)gmA * K + ac);  // +k0/4 later

    // B staging: thread -> col n (0..31), 4 consecutive k (strided scalar loads)
    const int bn   = t & 31;
    const int bk   = (t >> 5) * 4;              // k base in elements
    const float* Bcol = B + (size_t)bk * Nn + (n0 + bn);             // +k0*Nn later

    #pragma unroll 1
    for (int k0 = 0; k0 < K; k0 += 32) {
        // ---- stage A: 8 floats -> 4 packed u32 -> one ds_store_b128 ----
        {
            float4 v0 = make_float4(0.f, 0.f, 0.f, 0.f), v1 = v0;
            if (rowok) {
                const float4* p = Arow + (k0 >> 2);
                v0 = p[0];
                v1 = p[1];
            }
            uint4 pk;
            pk.x = pack_bf16x2(v0.x, v0.y);
            pk.y = pack_bf16x2(v0.z, v0.w);
            pk.z = pack_bf16x2(v1.x, v1.y);
            pk.w = pack_bf16x2(v1.z, v1.w);
            *(uint4*)&As[ar][(ac >> 1)] = pk;
        }
        // ---- stage B: 4 strided floats -> 2 packed u32 -> one ds_store_b64 ----
        {
            const float* p = Bcol + (size_t)k0 * Nn;
            float f0 = p[0];
            float f1 = p[(size_t)Nn];
            float f2 = p[(size_t)2 * Nn];
            float f3 = p[(size_t)3 * Nn];
            uint2 pk;
            pk.x = pack_bf16x2(f0, f1);
            pk.y = pack_bf16x2(f2, f3);
            *(uint2*)&Bs[bn][(bk >> 1)] = pk;
        }
        __syncthreads();

        // A fragment: lane holds K = {0..7}+8*half and {16..23}+8*half (16-bit 16x32 layout)
        const __bf16* ap = (const __bf16*)&As[wm * 16 + lm][0];
        v8bf alo = *(const v8bf*)(ap + 8 * half);
        v8bf ahi = *(const v8bf*)(ap + 16 + 8 * half);
        v16bf a = __builtin_shufflevector(alo, ahi, 0,1,2,3,4,5,6,7,8,9,10,11,12,13,14,15);

        // B fragment: lane holds K = {0..7}+8*half and {16..23}+8*half (matches A)
        const __bf16* bp = (const __bf16*)&Bs[wn * 16 + lm][0];
        v8bf blo = *(const v8bf*)(bp + 8 * half);
        v8bf bhi = *(const v8bf*)(bp + 16 + 8 * half);
        v16bf b = __builtin_shufflevector(blo, bhi, 0,1,2,3,4,5,6,7,8,9,10,11,12,13,14,15);

        acc = __builtin_amdgcn_wmma_f32_16x16x32_bf16(false, a, false, b,
                                                      (short)0, acc, false, false);
        __syncthreads();
    }

    const int gn = n0 + wn * 16 + lm;
    const float bv = bias[gn];
    #pragma unroll
    for (int r = 0; r < 8; ++r) {
        int gm = m0 + wm * 16 + r + 8 * half;   // C layout: VGPR r -> M = r + 8*half
        if (gm < M) {
            float v = acc[r] + bv;
            if (MODE == 1) v = fmaxf(v, 0.0f);
            if (MODE == 2) v += res[(size_t)gm * Nn + gn];
            C[(size_t)gm * Nn + gn] = v;
        }
    }
}

// ---------- misc elementwise ----------
__global__ void zero_kernel(float* p, long n) {
    long i = (long)blockIdx.x * blockDim.x + threadIdx.x;
    if (i < n) p[i] = 0.0f;
}

// ---------- edge phase (E' = E + N self loops), thread per (edge, head) ----------
__global__ void edge_score_kernel(const float* __restrict__ xl, const float* __restrict__ xr,
                                  const int* __restrict__ ei, const float* __restrict__ att,
                                  float* __restrict__ ebuf, unsigned* __restrict__ emax)
{
    const long EE = (long)NEDGES + NNODES;
    long tid = (long)blockIdx.x * blockDim.x + threadIdx.x;
    if (tid >= EE * NHEAD) return;
    int  hh = (int)(tid & 3);
    long e  = tid >> 2;
    int  s  = (e < NEDGES) ? ei[e]          : (int)(e - NEDGES);
    int  d  = (e < NEDGES) ? ei[NEDGES + e] : (int)(e - NEDGES);
    const float* pl = xl + (size_t)s * DMODEL + hh * CHEAD;
    const float* pr = xr + (size_t)d * DMODEL + hh * CHEAD;
    const float* pa = att + hh * CHEAD;
    float acc = 0.0f;
    #pragma unroll 4
    for (int c = 0; c < CHEAD; ++c) {
        float v = pl[c] + pr[c];
        v = (v > 0.0f) ? v : SLOPE * v;     // leaky relu
        acc += v * pa[c];
    }
    ebuf[tid] = acc;
    atomicMax(&emax[(size_t)d * NHEAD + hh], enc_f(acc));
}

__global__ void edge_exp_kernel(const int* __restrict__ ei, const unsigned* __restrict__ emax,
                                float* __restrict__ ebuf, float* __restrict__ denom)
{
    const long EE = (long)NEDGES + NNODES;
    long tid = (long)blockIdx.x * blockDim.x + threadIdx.x;
    if (tid >= EE * NHEAD) return;
    int  hh = (int)(tid & 3);
    long e  = tid >> 2;
    int  d  = (e < NEDGES) ? ei[NEDGES + e] : (int)(e - NEDGES);
    float ex = __expf(ebuf[tid] - dec_f(emax[(size_t)d * NHEAD + hh]));
    ebuf[tid] = ex;
    atomicAdd(&denom[(size_t)d * NHEAD + hh], ex);
}

__global__ void edge_accum_kernel(const float* __restrict__ xl, const int* __restrict__ ei,
                                  const float* __restrict__ ebuf, const float* __restrict__ denom,
                                  float* __restrict__ gat)
{
    const long EE = (long)NEDGES + NNODES;
    long tid = (long)blockIdx.x * blockDim.x + threadIdx.x;
    if (tid >= EE * NHEAD) return;
    int  hh = (int)(tid & 3);
    long e  = tid >> 2;
    int  s  = (e < NEDGES) ? ei[e]          : (int)(e - NEDGES);
    int  d  = (e < NEDGES) ? ei[NEDGES + e] : (int)(e - NEDGES);
    float alpha = ebuf[tid] / denom[(size_t)d * NHEAD + hh];
    const float* pl = xl  + (size_t)s * DMODEL + hh * CHEAD;
    float*       po = gat + (size_t)d * DMODEL + hh * CHEAD;
    #pragma unroll 4
    for (int c = 0; c < CHEAD; ++c) atomicAdd(&po[c], alpha * pl[c]);
}

__global__ void add_gat_kernel(float* __restrict__ h, const float* __restrict__ gat,
                               const float* __restrict__ gbias)
{
    int i = blockIdx.x * blockDim.x + threadIdx.x;
    if (i < NNODES * DMODEL) h[i] += gat[i] + gbias[i & (DMODEL - 1)];
}

// ---------- batch norm ----------
__global__ void bn_partial_kernel(const float* __restrict__ h,
                                  float* __restrict__ sums, float* __restrict__ sumsq)
{
    int d = threadIdx.x & (DMODEL - 1);
    int ro = threadIdx.x >> 7;                 // 0/1: two rows per block step
    float s = 0.0f, s2 = 0.0f;
    for (int r = blockIdx.x * 2 + ro; r < NNODES; r += gridDim.x * 2) {
        float v = h[(size_t)r * DMODEL + d];
        s += v; s2 += v * v;
    }
    atomicAdd(&sums[d], s);
    atomicAdd(&sumsq[d], s2);
}

__global__ void bn_finalize_kernel(const float* __restrict__ sums, const float* __restrict__ sumsq,
                                   float* __restrict__ mu, float* __restrict__ rs)
{
    int d = threadIdx.x;
    if (d < DMODEL) {
        float m  = sums[d] / (float)NNODES;
        float v  = sumsq[d] / (float)NNODES - m * m;   // biased variance
        mu[d] = m;
        rs[d] = rsqrtf(v + EPSBN);
    }
}

__global__ void bn_apply_kernel(float* __restrict__ h, const float* __restrict__ mu,
                                const float* __restrict__ rs, const float* __restrict__ g,
                                const float* __restrict__ b)
{
    int i = blockIdx.x * blockDim.x + threadIdx.x;
    if (i < NNODES * DMODEL) {
        int d = i & (DMODEL - 1);
        h[i] = (h[i] - mu[d]) * rs[d] * g[d] + b[d];
    }
}

// ---------- decoder: out[n] = h[n,:] @ decW + decb ----------
__global__ void decode_kernel(const float* __restrict__ h, const float* __restrict__ decW,
                              const float* __restrict__ decb, float* __restrict__ out)
{
    int n = blockIdx.x * blockDim.x + threadIdx.x;
    if (n >= NNODES) return;
    const float* row = h + (size_t)n * DMODEL;
    float a = 0.0f;
    #pragma unroll 8
    for (int d = 0; d < DMODEL; ++d) a += row[d] * decW[d];
    out[n] = a + decb[0];
}

// ---------- launcher ----------
extern "C" void kernel_launch(void* const* d_in, const int* in_sizes, int n_in,
                              void* d_out, int out_size, void* d_ws, size_t ws_size,
                              hipStream_t stream)
{
    const float* x     = (const float*)d_in[0];
    const int*   ei    = (const int*)  d_in[1];
    const float* embW  = (const float*)d_in[2];
    const float* embB  = (const float*)d_in[3];
    const float* Wl    = (const float*)d_in[4];
    const float* bl    = (const float*)d_in[5];
    const float* Wr    = (const float*)d_in[6];
    const float* br    = (const float*)d_in[7];
    const float* att   = (const float*)d_in[8];
    const float* gbias = (const float*)d_in[9];
    const float* bn1g  = (const float*)d_in[10];
    const float* bn1b  = (const float*)d_in[11];
    const float* ffW1  = (const float*)d_in[12];
    const float* ffb1  = (const float*)d_in[13];
    const float* ffW2  = (const float*)d_in[14];
    const float* ffb2  = (const float*)d_in[15];
    const float* bn2g  = (const float*)d_in[16];
    const float* bn2b  = (const float*)d_in[17];
    const float* decW  = (const float*)d_in[18];
    const float* decb  = (const float*)d_in[19];
    float* out = (float*)d_out;

    // workspace carve-out
    char* p = (char*)d_ws;
    auto take = [&](size_t bytes) -> char* {
        char* r = p; p += (bytes + 255) & ~(size_t)255; return r;
    };
    float*    h     = (float*)   take((size_t)NNODES * DMODEL * 4);
    float*    xl    = (float*)   take((size_t)NNODES * DMODEL * 4);
    float*    xr    = (float*)   take((size_t)NNODES * DMODEL * 4);
    float*    gat   = (float*)   take((size_t)NNODES * DMODEL * 4);
    float*    hid   = (float*)   take((size_t)NNODES * FHID   * 4);
    float*    ebuf  = (float*)   take(((size_t)NEDGES + NNODES) * NHEAD * 4);
    unsigned* emax  = (unsigned*)take((size_t)NNODES * NHEAD * 4);
    float*    denom = (float*)   take((size_t)NNODES * NHEAD * 4);
    float*    sums  = (float*)   take(DMODEL * 4);
    float*    sumsq = (float*)   take(DMODEL * 4);
    float*    mu    = (float*)   take(DMODEL * 4);
    float*    rs    = (float*)   take(DMODEL * 4);

    const long EEH    = ((long)NEDGES + NNODES) * NHEAD;          // 3.4M
    const int  eblk   = (int)((EEH + 255) / 256);
    const int  nd     = NNODES * DMODEL;
    const int  ndblk  = (nd + 255) / 256;
    const int  nhblk  = (NNODES * NHEAD + 255) / 256;

    dim3 blk(256);
    auto gemm_grid = [](int M, int Nn) { return dim3((unsigned)((M + 63) / 64), (unsigned)(Nn / 32)); };

    // embed: h = x @ embW + embB
    wmma_gemm_kernel<0, INDIM><<<gemm_grid(NNODES, DMODEL), blk, 0, stream>>>(
        x, embW, embB, nullptr, h, NNODES, DMODEL);

    for (int l = 0; l < NLAYER; ++l) {
        // projections
        wmma_gemm_kernel<0, DMODEL><<<gemm_grid(NNODES, DMODEL), blk, 0, stream>>>(
            h, Wl + (size_t)l * DMODEL * DMODEL, bl + l * DMODEL, nullptr, xl,
            NNODES, DMODEL);
        wmma_gemm_kernel<0, DMODEL><<<gemm_grid(NNODES, DMODEL), blk, 0, stream>>>(
            h, Wr + (size_t)l * DMODEL * DMODEL, br + l * DMODEL, nullptr, xr,
            NNODES, DMODEL);

        // clear accumulators
        zero_kernel<<<ndblk, blk, 0, stream>>>(gat, nd);
        zero_kernel<<<nhblk, blk, 0, stream>>>((float*)emax, NNODES * NHEAD);
        zero_kernel<<<nhblk, blk, 0, stream>>>(denom, NNODES * NHEAD);

        // segment softmax + aggregate
        edge_score_kernel<<<eblk, blk, 0, stream>>>(xl, xr, ei, att + l * NHEAD * CHEAD, ebuf, emax);
        edge_exp_kernel  <<<eblk, blk, 0, stream>>>(ei, emax, ebuf, denom);
        edge_accum_kernel<<<eblk, blk, 0, stream>>>(xl, ei, ebuf, denom, gat);

        // residual + gat bias, then BN1
        add_gat_kernel<<<ndblk, blk, 0, stream>>>(h, gat, gbias + l * DMODEL);
        zero_kernel<<<1, blk, 0, stream>>>(sums, DMODEL);
        zero_kernel<<<1, blk, 0, stream>>>(sumsq, DMODEL);
        bn_partial_kernel<<<512, blk, 0, stream>>>(h, sums, sumsq);
        bn_finalize_kernel<<<1, 128, 0, stream>>>(sums, sumsq, mu, rs);
        bn_apply_kernel<<<ndblk, blk, 0, stream>>>(h, mu, rs, bn1g + l * DMODEL, bn1b + l * DMODEL);

        // FF: hid = relu(h@W1+b1); h = hid@W2+b2 + h; then BN2
        wmma_gemm_kernel<1, DMODEL><<<gemm_grid(NNODES, FHID), blk, 0, stream>>>(
            h, ffW1 + (size_t)l * DMODEL * FHID, ffb1 + l * FHID, nullptr, hid,
            NNODES, FHID);
        wmma_gemm_kernel<2, FHID><<<gemm_grid(NNODES, DMODEL), blk, 0, stream>>>(
            hid, ffW2 + (size_t)l * FHID * DMODEL, ffb2 + l * DMODEL, h, h,
            NNODES, DMODEL);
        zero_kernel<<<1, blk, 0, stream>>>(sums, DMODEL);
        zero_kernel<<<1, blk, 0, stream>>>(sumsq, DMODEL);
        bn_partial_kernel<<<512, blk, 0, stream>>>(h, sums, sumsq);
        bn_finalize_kernel<<<1, 128, 0, stream>>>(sums, sumsq, mu, rs);
        bn_apply_kernel<<<ndblk, blk, 0, stream>>>(h, mu, rs, bn2g + l * DMODEL, bn2b + l * DMODEL);
    }

    // decode
    decode_kernel<<<(NNODES + 255) / 256, blk, 0, stream>>>(h, decW, decb, out);
}
